// ContrastiveLoss_60765197304292
// MI455X (gfx1250) — compile-verified, matrix operand order checked by reading, not run on hardware
//
#include <hip/hip_runtime.h>
#include <hip/hip_fp16.h>
#include <math.h>

#define N_NODES 8192
#define D_FEAT  256
#define NUM_NEG 64
#define TAU     0.5f
#define EPSL    1e-5f
#define MAXCLIP 20.0f
#define EXP_MAXCLIP 4.8516519540979028e8f  // exp(20)

typedef _Float16 h4  __attribute__((ext_vector_type(4)));
typedef _Float16 h8  __attribute__((ext_vector_type(8)));
typedef _Float16 h16 __attribute__((ext_vector_type(16)));
typedef float    f8  __attribute__((ext_vector_type(8)));

__device__ __forceinline__ float exp_stab(float sim) {
    float t = sim / (TAU + 1e-10f);
    t = fminf(fmaxf(t, -MAXCLIP), MAXCLIP);
    return fminf(expf(t), EXP_MAXCLIP);
}

__device__ __forceinline__ float dot4(const float4& a, const float4& b) {
    return a.x * b.x + a.y * b.y + a.z * b.z + a.w * b.w;
}

// ---------------------------------------------------------------------------
// Kernel 0: zero the scalar output accumulator (harness poisons d_out).
// ---------------------------------------------------------------------------
__global__ void cl_init_kernel(float* out) { out[0] = 0.0f; }

// ---------------------------------------------------------------------------
// Kernel 1: one wave (32 lanes) per node.
//   - compute ||x||, ||y||, x.y with f32 wave reductions
//   - write pos_sim_exp[n] = exp_stab((x.y)/(||x||*||y||))
//   - write z1 = x/max(||x||,1e-12) as row-major f16 table (for WMMA gathers)
// ---------------------------------------------------------------------------
__global__ void cl_norm_pos_kernel(const float* __restrict__ x,
                                   const float* __restrict__ y,
                                   _Float16* __restrict__ z1h,
                                   float* __restrict__ posbuf) {
    const int wid  = (blockIdx.x * blockDim.x + threadIdx.x) >> 5;  // node
    const int lane = threadIdx.x & 31;

    const float4* xr = (const float4*)(x + (size_t)wid * D_FEAT);
    const float4* yr = (const float4*)(y + (size_t)wid * D_FEAT);
    // each lane covers float4 slots {lane, lane+32}  (64 float4 = 256 floats)
    float4 xa = xr[lane], xb = xr[lane + 32];
    float4 ya = yr[lane], yb = yr[lane + 32];

    float sxx = dot4(xa, xa) + dot4(xb, xb);
    float syy = dot4(ya, ya) + dot4(yb, yb);
    float sxy = dot4(xa, ya) + dot4(xb, yb);

    #pragma unroll
    for (int off = 16; off > 0; off >>= 1) {
        sxx += __shfl_xor(sxx, off, 32);
        syy += __shfl_xor(syy, off, 32);
        sxy += __shfl_xor(sxy, off, 32);
    }

    const float invx = 1.0f / fmaxf(sqrtf(sxx), 1e-12f);
    const float invy = 1.0f / fmaxf(sqrtf(syy), 1e-12f);

    if (lane == 0) posbuf[wid] = exp_stab(sxy * invx * invy);

    _Float16* zrow = z1h + (size_t)wid * D_FEAT;
    h4 o0 = { (_Float16)(xa.x * invx), (_Float16)(xa.y * invx),
              (_Float16)(xa.z * invx), (_Float16)(xa.w * invx) };
    h4 o1 = { (_Float16)(xb.x * invx), (_Float16)(xb.y * invx),
              (_Float16)(xb.z * invx), (_Float16)(xb.w * invx) };
    ((h4*)zrow)[lane]       = o0;   // elements [lane*4 .. lane*4+3]
    ((h4*)(zrow + 128))[lane] = o1; // elements [128+lane*4 ..]
}

// ---------------------------------------------------------------------------
// Kernel 2: one wave per node; 4 WMMA groups of 16 negatives each,
// accumulating K=256 in 8 chunks of v_wmma_f32_16x16x32_f16.
//   A (16x32): 16 gathered negative rows (feature chunk)
//   B (32x16): z1[n] chunk replicated into all 16 columns
//   D[m][*]  : neg_m . z1[n]   (lanes 0-15 hold m=0..7, lanes 16-31 m=8..15)
// Per-lane K enumeration (identical for A and B):
//   lanes 0-15 : halves K{kb+0..7} then K{kb+16..23}
//   lanes 16-31: halves K{kb+8..15} then K{kb+24..31}
// ---------------------------------------------------------------------------
__global__ void cl_neg_loss_kernel(const int* __restrict__ idx,
                                   const _Float16* __restrict__ z1h,
                                   const float* __restrict__ posbuf,
                                   float* __restrict__ out) {
    const int n    = (blockIdx.x * blockDim.x + threadIdx.x) >> 5;  // node
    const int lane = threadIdx.x & 31;
    const int lsub = lane & 15;
    const int koff = (lane >> 4) ? 8 : 0;  // K sub-offset for this lane half

    const _Float16* zn = z1h + (size_t)n * D_FEAT;  // B source (replicated)
    float negacc = 0.0f;

    #pragma unroll
    for (int g = 0; g < 4; ++g) {
        const int j = idx[n * NUM_NEG + g * 16 + lsub];  // same neg for lane & lane+16
        const _Float16* za = z1h + (size_t)j * D_FEAT;

        f8 acc = {};
        #pragma unroll
        for (int c = 0; c < 8; ++c) {
            const int kb = c * 32 + koff;
            h8 a0 = *(const h8*)(za + kb);
            h8 a1 = *(const h8*)(za + kb + 16);
            h8 b0 = *(const h8*)(zn + kb);
            h8 b1 = *(const h8*)(zn + kb + 16);
            h16 A = __builtin_shufflevector(a0, a1, 0,1,2,3,4,5,6,7,8,9,10,11,12,13,14,15);
            h16 B = __builtin_shufflevector(b0, b1, 0,1,2,3,4,5,6,7,8,9,10,11,12,13,14,15);
            acc = __builtin_amdgcn_wmma_f32_16x16x32_f16(
                /*neg_a=*/false, A, /*neg_b=*/false, B,
                /*c_mod=*/(short)0, acc, /*reuse_a=*/false, /*reuse_b=*/false);
        }
        // lanes 0-15: acc[r] = dot for negative g*16+r   (r=0..7)
        // lanes 16-31: acc[r] = dot for negative g*16+8+r
        float s = 0.0f;
        #pragma unroll
        for (int r = 0; r < 8; ++r) s += exp_stab(acc[r]);
        negacc += s;
    }

    // combine the two half-wave partial sums (columns are replicated, so every
    // lane in a half holds the same value; xor-16 gives everyone the total)
    negacc += __shfl_xor(negacc, 16, 32);

    if (lane == 0) {
        const float pos  = posbuf[n];
        float loss = -logf(pos / (pos + negacc + EPSL));
        if (loss != loss) loss = 0.0f;  // NaN -> 0
        atomicAdd(out, loss * (1.0f / (float)N_NODES));
    }
}

// ---------------------------------------------------------------------------
extern "C" void kernel_launch(void* const* d_in, const int* in_sizes, int n_in,
                              void* d_out, int out_size, void* d_ws, size_t ws_size,
                              hipStream_t stream) {
    const float* x   = (const float*)d_in[0];
    const float* y   = (const float*)d_in[1];
    const int*   idx = (const int*)d_in[2];
    float*       out = (float*)d_out;

    _Float16* z1h    = (_Float16*)d_ws;                               // 4 MB
    float*    posbuf = (float*)((char*)d_ws + (size_t)N_NODES * D_FEAT * sizeof(_Float16));

    cl_init_kernel<<<1, 1, 0, stream>>>(out);

    // 8 waves per 256-thread block, one wave per node
    const int blocks = N_NODES / 8;
    cl_norm_pos_kernel<<<blocks, 256, 0, stream>>>(x, y, z1h, posbuf);
    cl_neg_loss_kernel<<<blocks, 256, 0, stream>>>(idx, z1h, posbuf, out);
}
